// InteractionBlock_51393578664115
// MI455X (gfx1250) — matrix operand order repaired and checked by reading, not compiled, not run on hardware
//
#include <hip/hip_runtime.h>
#include <hip/hip_bf16.h>
#include <math.h>

// ---------------- CDNA5 WMMA types ----------------
typedef __attribute__((ext_vector_type(16))) _Float16 v16h;
typedef __attribute__((ext_vector_type(8)))  _Float16 v8h;
typedef __attribute__((ext_vector_type(8)))  float    v8f;
typedef __attribute__((ext_vector_type(4)))  float    v4f;

// ---------------- problem constants ----------------
#define DIMC   768
#define NHEAD  6
#define DHEAD  128
#define NPTS   4
#define BATCH  8
#define HWX    1024            // 32*32 patch tokens
#define LCTX   5376            // 64*64 + 32*32 + 16*16
#define HIDF   192             // FFN hidden

// =====================================================================
// GEMM: out[M,N] = epilogue( A[M,K] @ W[K,N] + bias[N] )
//   epilogue: res==null -> acc+bias
//             res!=null -> res + (gamma?gamma[col]:1) * (acc+bias)
// fp32 inputs converted to f16 while staging into LDS (vectorized:
// 4x global_load_b128 -> v_cvt -> 2x ds_store_b128), f32 accumulate
// via v_wmma_f32_16x16x32_f16. Block tile 128(M) x 64(N), 256 threads
// = 8 waves, each wave owns a 32x32 sub-tile (2x2 WMMA accumulators).
// =====================================================================
__launch_bounds__(256)
__global__ void gemm_wmma_kernel(const float* __restrict__ A,
                                 const float* __restrict__ W,
                                 const float* __restrict__ bias,
                                 const float* __restrict__ res,
                                 const float* __restrict__ gamma,
                                 float* __restrict__ out,
                                 int M, int N, int K)
{
    __shared__ __align__(64) _Float16 As[128 * 32];   // [row][k]  8 KB
    __shared__ __align__(64) _Float16 Bs[64 * 32];    // [col][k]  4 KB (transposed)

    const int tid  = threadIdx.x;
    const int lane = tid & 31;
    const int wave = tid >> 5;          // 0..7
    const int wm   = wave & 3;          // 4 waves along M
    const int wn   = wave >> 2;         // 2 waves along N
    const int half = lane >> 4;         // hi/lo 16 lanes
    const int lr   = lane & 15;

    const int mBase = blockIdx.y * 128;
    const int nBase = blockIdx.x * 64;

    v8f acc[2][2] = {};

    // cooperative staging mapping
    const int ar  = tid >> 1;           // A row 0..127 (2 threads/row)
    const int ac0 = (tid & 1) * 16;     // A col start 0/16
    const int bn  = tid >> 2;           // B col 0..63 (4 threads/col)
    const int bk0 = (tid & 3) * 8;      // B k start 0/8/16/24

    const bool aOk = (mBase + ar) < M;
    const bool bOk = (nBase + bn) < N;
    const float* aSrcRow = A + (size_t)(mBase + ar) * K + ac0;

    for (int k0 = 0; k0 < K; k0 += 32) {
        // ---- stage A tile (fp32 -> f16), vectorized ----
        {
            _Float16* dst = &As[ar * 32 + ac0];
            if (aOk) {
                const float* src = aSrcRow + k0;
                if (k0 + 32 < K) __builtin_prefetch(src + 32, 0, 1); // global_prefetch_b8
                const v4f f0 = *(const v4f*)(src + 0);
                const v4f f1 = *(const v4f*)(src + 4);
                const v4f f2 = *(const v4f*)(src + 8);
                const v4f f3 = *(const v4f*)(src + 12);
                v8h h0, h1;
                #pragma unroll
                for (int i = 0; i < 4; ++i) {
                    h0[i]     = (_Float16)f0[i];
                    h0[4 + i] = (_Float16)f1[i];
                    h1[i]     = (_Float16)f2[i];
                    h1[4 + i] = (_Float16)f3[i];
                }
                *(v8h*)(dst)     = h0;   // ds_store_b128
                *(v8h*)(dst + 8) = h1;   // ds_store_b128
            } else {
                const v8h z = {};
                *(v8h*)(dst)     = z;
                *(v8h*)(dst + 8) = z;
            }
        }
        // ---- stage B tile transposed (fp32 -> f16), one b128 store ----
        {
            v8h hb = {};
            if (bOk) {
                const float* wp = W + (size_t)(k0 + bk0) * N + (nBase + bn);
                #pragma unroll
                for (int i = 0; i < 8; ++i)
                    hb[i] = (_Float16)wp[(size_t)i * N];
            }
            *(v8h*)&Bs[bn * 32 + bk0] = hb;  // ds_store_b128
        }
        __syncthreads();

        // ---- build fragments (A 16x32 / B 32x16, contiguous k = 16*half + i) ----
        v16h afrag[2], bfrag[2];
        #pragma unroll
        for (int s = 0; s < 2; ++s) {
            const int row = wm * 32 + s * 16 + lr;
            const v8h a0 = *(const v8h*)&As[row * 32 + half * 16];       // K 0-7 / 16-23
            const v8h a1 = *(const v8h*)&As[row * 32 + half * 16 + 8];   // K 8-15 / 24-31
            v16h a;
            #pragma unroll
            for (int i = 0; i < 8; ++i) { a[i] = a0[i]; a[8 + i] = a1[i]; }
            afrag[s] = a;
        }
        #pragma unroll
        for (int t = 0; t < 2; ++t) {
            const int col = wn * 32 + t * 16 + lr;
            bfrag[t] = *(const v16h*)&Bs[col * 32 + half * 16];          // K 0-15 / 16-31
        }

        #pragma unroll
        for (int s = 0; s < 2; ++s)
            #pragma unroll
            for (int t = 0; t < 2; ++t)
                acc[s][t] = __builtin_amdgcn_wmma_f32_16x16x32_f16(
                    false, afrag[s], false, bfrag[t],
                    (short)0, acc[s][t], false, false);
        __syncthreads();
    }

    // ---- epilogue: C/D layout row = half*8+v, col = lane&15 ----
    #pragma unroll
    for (int s = 0; s < 2; ++s) {
        #pragma unroll
        for (int t = 0; t < 2; ++t) {
            const int col = nBase + wn * 32 + t * 16 + lr;
            if (col >= N) continue;
            const float bcol = bias[col];
            const float gcol = gamma ? gamma[col] : 1.0f;
            #pragma unroll
            for (int vv = 0; vv < 8; ++vv) {
                const int row = mBase + wm * 32 + s * 16 + half * 8 + vv;
                if (row >= M) continue;
                float val = acc[s][t][vv] + bcol;
                if (res) val = res[(size_t)row * N + col] + gcol * val;
                out[(size_t)row * N + col] = val;
            }
        }
    }
}

// =====================================================================
// LayerNorm over last dim (768). One block per row.
// =====================================================================
__launch_bounds__(256)
__global__ void ln_kernel(const float* __restrict__ x,
                          const float* __restrict__ g,
                          const float* __restrict__ b,
                          float* __restrict__ out)
{
    const size_t row = blockIdx.x;
    const float* xr = x + row * DIMC;
    float s = 0.f, ss = 0.f;
    for (int i = threadIdx.x; i < DIMC; i += 256) {
        const float v = xr[i];
        s += v; ss += v * v;
    }
    __shared__ float rs[256], rss[256];
    rs[threadIdx.x] = s; rss[threadIdx.x] = ss;
    __syncthreads();
    for (int off = 128; off > 0; off >>= 1) {
        if (threadIdx.x < off) {
            rs[threadIdx.x]  += rs[threadIdx.x + off];
            rss[threadIdx.x] += rss[threadIdx.x + off];
        }
        __syncthreads();
    }
    const float m   = rs[0]  * (1.0f / DIMC);
    const float var = rss[0] * (1.0f / DIMC) - m * m;
    const float inv = rsqrtf(var + 1e-6f);
    for (int i = threadIdx.x; i < DIMC; i += 256)
        out[row * DIMC + i] = (xr[i] - m) * inv * g[i] + b[i];
}

// =====================================================================
// In-place softmax over groups of glen (12 or 4); 1 thread per group.
// =====================================================================
__launch_bounds__(256)
__global__ void softmax_kernel(float* __restrict__ a, int ngroups, int glen)
{
    const int gi = blockIdx.x * 256 + threadIdx.x;
    if (gi >= ngroups) return;
    float* p = a + (size_t)gi * glen;
    float mx = -1e30f;
    for (int i = 0; i < glen; ++i) mx = fmaxf(mx, p[i]);
    float sum = 0.f;
    for (int i = 0; i < glen; ++i) { const float e = expf(p[i] - mx); p[i] = e; sum += e; }
    const float inv = 1.0f / sum;
    for (int i = 0; i < glen; ++i) p[i] *= inv;
}

// =====================================================================
// MSDA bilinear sampling + attention-weighted reduce:
//   out[bq, d] = sum_{l,p} aw[bq,h,l,p] * bilinear(v_level_l, loc)
// One block per (b,q); 256 threads cover the 768 channels.
// =====================================================================
struct MsdaCfg {
    int Lq, Lv, L, refStride;
    int Hh[3], Ww[3], start[3];
};

__launch_bounds__(256)
__global__ void msda_sample_kernel(const float* __restrict__ v,
                                   const float* __restrict__ ref,
                                   const float* __restrict__ off,
                                   const float* __restrict__ aw,
                                   float* __restrict__ outp,
                                   MsdaCfg cfg)
{
    const int bq = blockIdx.x;                 // b*Lq + q
    const int b  = bq / cfg.Lq;
    const float* refp = ref + (size_t)bq * cfg.refStride * 2;

    for (int d = threadIdx.x; d < DIMC; d += 256) {
        const int h = d >> 7;                  // head = d / 128
        const float* offp = off + ((size_t)bq * NHEAD + h) * cfg.L * NPTS * 2;
        const float* awp  = aw  + ((size_t)bq * NHEAD + h) * cfg.L * NPTS;
        float acc = 0.f;
        for (int l = 0; l < cfg.L; ++l) {
            const int Hl = cfg.Hh[l], Wl = cfg.Ww[l];
            const int rl = (cfg.refStride == 1) ? 0 : l;
            const float rx = refp[rl * 2 + 0];
            const float ry = refp[rl * 2 + 1];
            const float* base = v + ((size_t)b * cfg.Lv + cfg.start[l]) * DIMC + d;
            for (int p = 0; p < NPTS; ++p) {
                const float ox = offp[(l * NPTS + p) * 2 + 0];
                const float oy = offp[(l * NPTS + p) * 2 + 1];
                const float w  = awp[l * NPTS + p];
                // loc = ref + off/[Wl,Hl]; x = loc_x*Wl - 0.5 = ref_x*Wl + off_x - 0.5
                const float xx = rx * (float)Wl + ox - 0.5f;
                const float yy = ry * (float)Hl + oy - 0.5f;
                const float x0f = floorf(xx), y0f = floorf(yy);
                const float lx = xx - x0f, ly = yy - y0f;
                const int x0 = (int)x0f, y0 = (int)y0f;
                const int x1 = x0 + 1,   y1 = y0 + 1;
                const bool vx0 = (x0 >= 0) & (x0 < Wl);
                const bool vx1 = (x1 >= 0) & (x1 < Wl);
                const bool vy0 = (y0 >= 0) & (y0 < Hl);
                const bool vy1 = (y1 >= 0) & (y1 < Hl);
                const float g00 = (vx0 & vy0) ? base[(size_t)(y0 * Wl + x0) * DIMC] : 0.f;
                const float g01 = (vx1 & vy0) ? base[(size_t)(y0 * Wl + x1) * DIMC] : 0.f;
                const float g10 = (vx0 & vy1) ? base[(size_t)(y1 * Wl + x0) * DIMC] : 0.f;
                const float g11 = (vx1 & vy1) ? base[(size_t)(y1 * Wl + x1) * DIMC] : 0.f;
                acc += w * (g00 * (1.f - lx) * (1.f - ly) + g01 * lx * (1.f - ly) +
                            g10 * (1.f - lx) * ly          + g11 * lx * ly);
            }
        }
        outp[(size_t)bq * DIMC + d] = acc;
    }
}

// =====================================================================
// Depthwise 3x3 SAME conv over the three pyramid segments + exact GELU.
// One thread per (b, row, channel).
// =====================================================================
__launch_bounds__(256)
__global__ void dwgelu_kernel(const float* __restrict__ h1,
                              const float* __restrict__ kw,
                              const float* __restrict__ kb,
                              float* __restrict__ outp,
                              int total)
{
    const int idx = blockIdx.x * 256 + threadIdx.x;
    if (idx >= total) return;
    const int ch   = idx % HIDF;
    const int rowg = idx / HIDF;              // b*5376 + row
    const int b    = rowg / LCTX;
    const int row  = rowg % LCTX;
    int Hs, Ws, base;
    if      (row < 4096) { Hs = 64; Ws = 64; base = 0;    }
    else if (row < 5120) { Hs = 32; Ws = 32; base = 4096; }
    else                 { Hs = 16; Ws = 16; base = 5120; }
    const int r = row - base;
    const int y = r / Ws, x = r % Ws;
    float acc = kb[ch];
    #pragma unroll
    for (int dy = 0; dy < 3; ++dy) {
        const int yy = y + dy - 1;
        if (yy < 0 || yy >= Hs) continue;
        #pragma unroll
        for (int dx = 0; dx < 3; ++dx) {
            const int xxp = x + dx - 1;
            if (xxp < 0 || xxp >= Ws) continue;
            acc += h1[((size_t)b * LCTX + base + yy * Ws + xxp) * HIDF + ch]
                 * kw[(dy * 3 + dx) * HIDF + ch];
        }
    }
    outp[idx] = 0.5f * acc * (1.0f + erff(acc * 0.70710678118654752f));
}

// =====================================================================
// Host-side orchestration
// =====================================================================
static inline void launch_gemm(const float* A, const float* W, const float* bias,
                               const float* res, const float* gamma, float* out,
                               int M, int N, int K, hipStream_t s)
{
    dim3 grid((N + 63) / 64, (M + 127) / 128);
    gemm_wmma_kernel<<<grid, 256, 0, s>>>(A, W, bias, res, gamma, out, M, N, K);
}

extern "C" void kernel_launch(void* const* d_in, const int* in_sizes, int n_in,
                              void* d_out, int out_size, void* d_ws, size_t ws_size,
                              hipStream_t stream)
{
    (void)in_sizes; (void)n_in; (void)out_size; (void)ws_size;

    // input order: x, c, ref1, ref2, then params in insertion order
    const float* x    = (const float*)d_in[0];
    const float* c    = (const float*)d_in[1];
    const float* ref1 = (const float*)d_in[2];
    const float* ref2 = (const float*)d_in[3];
    const float* injqn_g = (const float*)d_in[4];
    const float* injqn_b = (const float*)d_in[5];
    const float* injfn_g = (const float*)d_in[6];
    const float* injfn_b = (const float*)d_in[7];
    const float* Woff1 = (const float*)d_in[8];
    const float* boff1 = (const float*)d_in[9];
    const float* Wa1   = (const float*)d_in[10];
    const float* ba1   = (const float*)d_in[11];
    const float* Wv1   = (const float*)d_in[12];
    const float* bv1   = (const float*)d_in[13];
    const float* Wo1   = (const float*)d_in[14];
    const float* bo1   = (const float*)d_in[15];
    const float* gamma = (const float*)d_in[16];
    const float* extqn_g = (const float*)d_in[17];
    const float* extqn_b = (const float*)d_in[18];
    const float* extfn_g = (const float*)d_in[19];
    const float* extfn_b = (const float*)d_in[20];
    const float* ffnn_g  = (const float*)d_in[21];
    const float* ffnn_b  = (const float*)d_in[22];
    const float* Woff2 = (const float*)d_in[23];
    const float* boff2 = (const float*)d_in[24];
    const float* Wa2   = (const float*)d_in[25];
    const float* ba2   = (const float*)d_in[26];
    const float* Wv2   = (const float*)d_in[27];
    const float* bv2   = (const float*)d_in[28];
    const float* Wo2   = (const float*)d_in[29];
    const float* bo2   = (const float*)d_in[30];
    const float* W1    = (const float*)d_in[31];
    const float* b1    = (const float*)d_in[32];
    const float* dwk   = (const float*)d_in[33];
    const float* dwb   = (const float*)d_in[34];
    const float* W2    = (const float*)d_in[35];
    const float* b2    = (const float*)d_in[36];

    const int rowsX = BATCH * HWX;    // 8192
    const int rowsC = BATCH * LCTX;   // 43008

    float* ws   = (float*)d_ws;
    float* BIG0 = ws;                          // 33,030,144 : fn1 -> qn2 -> ffn_in
    float* BIG1 = ws + 33030144;               // 33,030,144 : v1 -> attnin2
    float* MED0 = ws + 66060288;               //  6,291,456 : qn1 -> fn2
    float* MED1 = ws + 72351744;               //  6,291,456 : attnin1 -> v2
    float* H1B  = ws + 78643200;               //  8,257,536 : h1
    float* H2B  = ws + 86900736;               //  8,257,536 : h1dw
    float* OFF1 = ws + 95158272;               //  1,179,648
    float* AW1  = ws + 96337920;               //    589,824
    float* OFF2 = ws + 96927744;               //  2,064,384
    float* AW2  = ws + 98992128;               //  1,032,192

    float* outX = (float*)d_out;               // (8,1024,768)
    float* outC = outX + (size_t)rowsX * DIMC; // (8,5376,768)

    // ---------------- stage 1: inject attention (q = x patches, feat = c)
    ln_kernel<<<rowsX, 256, 0, stream>>>(x, injqn_g, injqn_b, MED0);   // qn1
    ln_kernel<<<rowsC, 256, 0, stream>>>(c, injfn_g, injfn_b, BIG0);   // fn1
    launch_gemm(BIG0, Wv1, bv1, nullptr, nullptr, BIG1, rowsC, DIMC, DIMC, stream);          // v1
    launch_gemm(MED0, Woff1, boff1, nullptr, nullptr, OFF1, rowsX, NHEAD*3*NPTS*2, DIMC, stream);
    launch_gemm(MED0, Wa1,   ba1,   nullptr, nullptr, AW1,  rowsX, NHEAD*3*NPTS,   DIMC, stream);
    {
        const int ng = rowsX * NHEAD;          // softmax over L*P = 12
        softmax_kernel<<<(ng + 255) / 256, 256, 0, stream>>>(AW1, ng, 3 * NPTS);
    }
    {
        MsdaCfg cfg;
        cfg.Lq = HWX; cfg.Lv = LCTX; cfg.L = 3; cfg.refStride = 3;
        cfg.Hh[0] = 64; cfg.Hh[1] = 32; cfg.Hh[2] = 16;
        cfg.Ww[0] = 64; cfg.Ww[1] = 32; cfg.Ww[2] = 16;
        cfg.start[0] = 0; cfg.start[1] = 4096; cfg.start[2] = 5120;
        msda_sample_kernel<<<rowsX, 256, 0, stream>>>(BIG1, ref1, OFF1, AW1, MED1, cfg);
    }
    // xp_new = x + gamma * (attnin1 @ Wo1 + bo1)   -> d_out x region
    launch_gemm(MED1, Wo1, bo1, x, gamma, outX, rowsX, DIMC, DIMC, stream);

    // ---------------- stage 2: extract attention (q = c, feat = xp_new)
    ln_kernel<<<rowsC, 256, 0, stream>>>(c,    extqn_g, extqn_b, BIG0); // qn2
    ln_kernel<<<rowsX, 256, 0, stream>>>(outX, extfn_g, extfn_b, MED0); // fn2
    launch_gemm(MED0, Wv2, bv2, nullptr, nullptr, MED1, rowsX, DIMC, DIMC, stream);          // v2
    launch_gemm(BIG0, Woff2, boff2, nullptr, nullptr, OFF2, rowsC, NHEAD*1*NPTS*2, DIMC, stream);
    launch_gemm(BIG0, Wa2,   ba2,   nullptr, nullptr, AW2,  rowsC, NHEAD*1*NPTS,   DIMC, stream);
    {
        const int ng = rowsC * NHEAD;          // softmax over L*P = 4
        softmax_kernel<<<(ng + 255) / 256, 256, 0, stream>>>(AW2, ng, NPTS);
    }
    {
        MsdaCfg cfg;
        cfg.Lq = LCTX; cfg.Lv = HWX; cfg.L = 1; cfg.refStride = 1;
        cfg.Hh[0] = 32; cfg.Hh[1] = 0; cfg.Hh[2] = 0;
        cfg.Ww[0] = 32; cfg.Ww[1] = 0; cfg.Ww[2] = 0;
        cfg.start[0] = 0; cfg.start[1] = 0; cfg.start[2] = 0;
        msda_sample_kernel<<<rowsC, 256, 0, stream>>>(MED1, ref2, OFF2, AW2, BIG1, cfg);
    }
    // c1 = c + (attnin2 @ Wo2 + bo2)   -> d_out c region
    launch_gemm(BIG1, Wo2, bo2, c, nullptr, outC, rowsC, DIMC, DIMC, stream);

    // ---------------- conv-FFN on c1
    ln_kernel<<<rowsC, 256, 0, stream>>>(outC, ffnn_g, ffnn_b, BIG0);   // ffn_in
    launch_gemm(BIG0, W1, b1, nullptr, nullptr, H1B, rowsC, HIDF, DIMC, stream);             // h1
    {
        const int total = rowsC * HIDF;
        dwgelu_kernel<<<(total + 255) / 256, 256, 0, stream>>>(H1B, dwk, dwb, H2B, total);
    }
    // c_out = c1 + (h1dw @ W2 + b2)   (in-place accumulate into d_out c region)
    launch_gemm(H2B, W2, b2, outC, nullptr, outC, rowsC, DIMC, HIDF, stream);
}
